// DINScorer_35450660061554
// MI455X (gfx1250) — matrix operand order repaired and checked by reading, not compiled, not run on hardware
//
#include <hip/hip_runtime.h>
#include <hip/hip_bf16.h>

// ---------------------------------------------------------------------------
// DIN scorer for MI455X (gfx1250), bf16 WMMA (16x16x32) with f32 accumulate.
// B=2048, L=200, D=128.
// Round 2: prevent B-fragment spill-to-scratch by limiting unrolling of the
// N-tile loops (#pragma unroll 1); keep K-chains fully unrolled.
// ---------------------------------------------------------------------------

typedef __attribute__((ext_vector_type(16))) __bf16 v16bf;
typedef __attribute__((ext_vector_type(8)))  __bf16 v8bf;
typedef __attribute__((ext_vector_type(8)))  float  v8f;

#define B_DIM 2048
#define L_DIM 200
#define D_DIM 128

#define V8F_ZERO {0.f,0.f,0.f,0.f,0.f,0.f,0.f,0.f}

__device__ __forceinline__ v8f wmma_bf16f32(v16bf a, v16bf b, v8f c) {
  return __builtin_amdgcn_wmma_f32_16x16x32_bf16(false, a, false, b, (short)0, c,
                                                 false, false);
}

// A-matrix fragment (16x32 bf16), row-major LDS tile with row stride ldk.
// lane: M = lane&15; kbase = (lane&16)?8:0; elem i: K = kbase + (i<8 ? i : 16+i-8)
__device__ __forceinline__ v16bf load_a_frag(const __bf16* tile, int ldk, int k0, int lane) {
  const int m  = lane & 15;
  const int kb = (lane & 16) ? 8 : 0;
  const __bf16* p = tile + m * ldk + k0 + kb;
  union { v16bf v; v8bf h[2]; } u;
  u.h[0] = *(const v8bf*)(p);
  u.h[1] = *(const v8bf*)(p + 16);
  return u.v;
}

// B-matrix fragment from pre-packed global layout [tileIdx][lane][16] bf16:
// one contiguous 32B load per lane.
__device__ __forceinline__ v16bf load_b_frag(const __bf16* pk, int tileIdx, int lane) {
  return *(const v16bf*)(pk + ((size_t)tileIdx * 32 + lane) * 16);
}

__device__ __forceinline__ float sigf(float x) { return 1.0f / (1.0f + __expf(-x)); }

// ---------------------------------------------------------------------------
// Weight pre-pack: f32 row-major (K x N) -> bf16 B-fragment layout
// dst[((nt*(K/32)+ks)*32 + lane)*16 + i] = src[(ks*32 + ((lane>>4)&1)*8 + (i<8 ? i : 8+i))*N + nt*16 + (lane&15)]
// ---------------------------------------------------------------------------
__global__ void pack_b_kernel(const float* __restrict__ src, __bf16* __restrict__ dst,
                              int K, int N, int total) {
  int idx = blockIdx.x * blockDim.x + threadIdx.x;
  if (idx >= total) return;
  int i     = idx & 15;
  int lane  = (idx >> 4) & 31;
  int rest  = idx >> 9;
  int kst   = K >> 5;
  int ks    = rest % kst;
  int nt    = rest / kst;
  int k = ks * 32 + ((lane >> 4) & 1) * 8 + (i < 8 ? i : 8 + i);
  int n = nt * 16 + (lane & 15);
  dst[idx] = (__bf16)src[(size_t)k * N + n];
}

// ---------------------------------------------------------------------------
// Stage 1: attention MLP + weighted pooling. One workgroup (8 waves) per batch b.
// ---------------------------------------------------------------------------
__global__ __launch_bounds__(256) void din_attn_kernel(
    const int* __restrict__ in_item_id, const int* __restrict__ item_id,
    const float* __restrict__ item_emb,
    const __bf16* __restrict__ aw1p, const float* __restrict__ ab1,
    const __bf16* __restrict__ aw2p, const float* __restrict__ ab2,
    const float* __restrict__ afw, const float* __restrict__ afb,
    float* __restrict__ attn_out)
{
  __shared__ float  tt[128];
  __shared__ float  attns[128];
  __shared__ float  wscore[8][16];
  __shared__ float  seqtile[8][16][128];
  __shared__ __bf16 keytile[8][16][384];
  __shared__ __bf16 h1tile[8][16][128];

  const int b    = blockIdx.x;
  const int tid  = threadIdx.x;
  const int wave = tid >> 5;
  const int lane = tid & 31;
  const int m    = lane & 15;
  const int half = lane >> 4;
  const int nn   = lane & 15;
  const int hi   = (lane & 16) ? 8 : 0;

  if (tid < 128) {
    tt[tid]    = item_emb[(size_t)item_id[b] * D_DIM + tid];
    attns[tid] = 0.0f;
  }
  __syncthreads();

  const float afbv = afb[0];
  float attnacc[4] = {0.f, 0.f, 0.f, 0.f};

#pragma unroll 1
  for (int it = 0; it < 2; ++it) {
    const int tile = it * 8 + wave;            // 0..15 (rows >= 200 are masked out)
    const int l    = tile * 16 + m;
    const int id   = (l < L_DIM) ? in_item_id[(size_t)b * L_DIM + l] : 0;  // row 0 of table is zero
    const float* srow = item_emb + (size_t)id * D_DIM;
    const int dbase = half * 64;

    // Build key tile: [t | t*s | t-s] (bf16) and keep s (f32) for pooling.
#pragma unroll
    for (int j = 0; j < 16; ++j) {
      const int d = dbase + j * 4;
      const float4 s4 = *(const float4*)(srow + d);
      *(float4*)&seqtile[wave][m][d] = s4;
      const float sv[4] = { s4.x, s4.y, s4.z, s4.w };
#pragma unroll
      for (int c = 0; c < 4; ++c) {
        const float tv = tt[d + c];
        keytile[wave][m][d + c]       = (__bf16)tv;
        keytile[wave][m][128 + d + c] = (__bf16)(tv * sv[c]);
        keytile[wave][m][256 + d + c] = (__bf16)(tv - sv[c]);
      }
    }
    __syncthreads();

    // GEMM1: (16x384)@(384x128) + ab1, sigmoid -> h1tile
#pragma unroll 1
    for (int nt = 0; nt < 8; ++nt) {
      v8f acc = V8F_ZERO;
#pragma unroll
      for (int ks = 0; ks < 12; ++ks) {
        v16bf a  = load_a_frag(&keytile[wave][0][0], 384, ks * 32, lane);
        v16bf bf = load_b_frag(aw1p, nt * 12 + ks, lane);
        acc = wmma_bf16f32(a, bf, acc);
      }
      const float bias = ab1[nt * 16 + nn];
#pragma unroll
      for (int v = 0; v < 8; ++v)
        h1tile[wave][v + hi][nt * 16 + nn] = (__bf16)sigf(acc[v] + bias);
    }
    __syncthreads();

    // GEMM2: (16x128)@(128x64) + ab2, sigmoid, dot with afw -> per-row score
    float partial[8] = {0.f,0.f,0.f,0.f,0.f,0.f,0.f,0.f};
#pragma unroll 1
    for (int nt = 0; nt < 4; ++nt) {
      v8f acc = V8F_ZERO;
#pragma unroll
      for (int ks = 0; ks < 4; ++ks) {
        v16bf a  = load_a_frag(&h1tile[wave][0][0], 128, ks * 32, lane);
        v16bf bf = load_b_frag(aw2p, nt * 4 + ks, lane);
        acc = wmma_bf16f32(a, bf, acc);
      }
      const float bias = ab2[nt * 16 + nn];
      const float w    = afw[nt * 16 + nn];
#pragma unroll
      for (int v = 0; v < 8; ++v)
        partial[v] += sigf(acc[v] + bias) * w;
    }
#pragma unroll
    for (int v = 0; v < 8; ++v) {
      partial[v] += __shfl_xor(partial[v], 8, 32);
      partial[v] += __shfl_xor(partial[v], 4, 32);
      partial[v] += __shfl_xor(partial[v], 2, 32);
      partial[v] += __shfl_xor(partial[v], 1, 32);
    }
    if (nn == 0) {
#pragma unroll
      for (int v = 0; v < 8; ++v) {
        const int mm = v + hi;
        const int ll = tile * 16 + mm;
        float sc = partial[v] + afbv;
        if (ll >= L_DIM || in_item_id[(size_t)b * L_DIM + ll] == 0) sc = 0.f;
        wscore[wave][mm] = sc;
      }
    }
    __syncthreads();

    // attn += score * seq   (each lane owns 4 consecutive d's)
#pragma unroll
    for (int mm = 0; mm < 16; ++mm) {
      const float sc  = wscore[wave][mm];
      const float4 s4 = *(const float4*)&seqtile[wave][mm][lane * 4];
      attnacc[0] += sc * s4.x; attnacc[1] += sc * s4.y;
      attnacc[2] += sc * s4.z; attnacc[3] += sc * s4.w;
    }
    __syncthreads();
  }

#pragma unroll
  for (int j = 0; j < 4; ++j) atomicAdd(&attns[lane * 4 + j], attnacc[j]);
  __syncthreads();
  if (tid < 128) attn_out[(size_t)b * D_DIM + tid] = attns[tid];
}

// ---------------------------------------------------------------------------
// Stage 2: per-batch DNN. 4 waves per block, each wave one 16-row batch tile.
// ---------------------------------------------------------------------------
__global__ __launch_bounds__(128) void din_dnn_kernel(
    const float* __restrict__ attn_in, const int* __restrict__ item_id,
    const float* __restrict__ item_emb, const float* __restrict__ item_bias_tab,
    const __bf16* __restrict__ nwp, const float* __restrict__ nb,
    const __bf16* __restrict__ dw1p, const float* __restrict__ db1,
    const __bf16* __restrict__ dw2p, const float* __restrict__ db2,
    const __bf16* __restrict__ dw3p, const float* __restrict__ db3,
    const float* __restrict__ fw, const float* __restrict__ fb,
    float* __restrict__ out)
{
  __shared__ __bf16 aT[4][16][128];
  __shared__ float  tT[4][16][128];
  __shared__ __bf16 catT[4][16][384];
  __shared__ __bf16 g1T[4][16][512];
  __shared__ __bf16 g2T[4][16][256];

  const int tid  = threadIdx.x;
  const int wave = tid >> 5;
  const int lane = tid & 31;
  const int tile = blockIdx.x * 4 + wave;      // 0..127
  const int r0   = tile * 16;
  const int m    = lane & 15;
  const int half = lane >> 4;
  const int nn   = lane & 15;
  const int hi   = (lane & 16) ? 8 : 0;

  // P0: stage attn rows (bf16 A tile), target rows (f32 + into cat middle block)
  {
    const int r = r0 + m;
    const float* arow = attn_in + (size_t)r * D_DIM;
    const float* trow = item_emb + (size_t)item_id[r] * D_DIM;
    const int dbase = half * 64;
#pragma unroll
    for (int j = 0; j < 16; ++j) {
      const int d = dbase + j * 4;
      const float4 a4 = *(const float4*)(arow + d);
      const float4 t4 = *(const float4*)(trow + d);
      aT[wave][m][d]     = (__bf16)a4.x; aT[wave][m][d + 1] = (__bf16)a4.y;
      aT[wave][m][d + 2] = (__bf16)a4.z; aT[wave][m][d + 3] = (__bf16)a4.w;
      *(float4*)&tT[wave][m][d] = t4;
      catT[wave][m][128 + d]     = (__bf16)t4.x; catT[wave][m][128 + d + 1] = (__bf16)t4.y;
      catT[wave][m][128 + d + 2] = (__bf16)t4.z; catT[wave][m][128 + d + 3] = (__bf16)t4.w;
    }
  }
  __syncthreads();

  // P1: attn2 = a @ nw + nb ; cat = [attn2, t, t*attn2]
#pragma unroll 1
  for (int nt = 0; nt < 8; ++nt) {
    v8f acc = V8F_ZERO;
#pragma unroll
    for (int ks = 0; ks < 4; ++ks) {
      v16bf a  = load_a_frag(&aT[wave][0][0], 128, ks * 32, lane);
      v16bf bf = load_b_frag(nwp, nt * 4 + ks, lane);
      acc = wmma_bf16f32(a, bf, acc);
    }
    const float bias = nb[nt * 16 + nn];
#pragma unroll
    for (int v = 0; v < 8; ++v) {
      const int mm = v + hi;
      const float x = acc[v] + bias;
      catT[wave][mm][nt * 16 + nn]       = (__bf16)x;
      catT[wave][mm][256 + nt * 16 + nn] = (__bf16)(x * tT[wave][mm][nt * 16 + nn]);
    }
  }
  __syncthreads();

  // P2: g1 = sig(cat @ dw1 + db1)   (16x384)@(384x512)
#pragma unroll 1
  for (int nt = 0; nt < 32; ++nt) {
    v8f acc = V8F_ZERO;
#pragma unroll
    for (int ks = 0; ks < 12; ++ks) {
      v16bf a  = load_a_frag(&catT[wave][0][0], 384, ks * 32, lane);
      v16bf bf = load_b_frag(dw1p, nt * 12 + ks, lane);
      acc = wmma_bf16f32(a, bf, acc);
    }
    const float bias = db1[nt * 16 + nn];
#pragma unroll
    for (int v = 0; v < 8; ++v)
      g1T[wave][v + hi][nt * 16 + nn] = (__bf16)sigf(acc[v] + bias);
  }
  __syncthreads();

  // P3: g2 = sig(g1 @ dw2 + db2)   (16x512)@(512x256)
#pragma unroll 1
  for (int nt = 0; nt < 16; ++nt) {
    v8f acc = V8F_ZERO;
#pragma unroll
    for (int ks = 0; ks < 16; ++ks) {
      v16bf a  = load_a_frag(&g1T[wave][0][0], 512, ks * 32, lane);
      v16bf bf = load_b_frag(dw2p, nt * 16 + ks, lane);
      acc = wmma_bf16f32(a, bf, acc);
    }
    const float bias = db2[nt * 16 + nn];
#pragma unroll
    for (int v = 0; v < 8; ++v)
      g2T[wave][v + hi][nt * 16 + nn] = (__bf16)sigf(acc[v] + bias);
  }
  __syncthreads();

  // P4: g3 = sig(g2 @ dw3 + db3); out = g3 @ fw + fb + item_bias
  float partial[8] = {0.f,0.f,0.f,0.f,0.f,0.f,0.f,0.f};
#pragma unroll 1
  for (int nt = 0; nt < 8; ++nt) {
    v8f acc = V8F_ZERO;
#pragma unroll
    for (int ks = 0; ks < 8; ++ks) {
      v16bf a  = load_a_frag(&g2T[wave][0][0], 256, ks * 32, lane);
      v16bf bf = load_b_frag(dw3p, nt * 8 + ks, lane);
      acc = wmma_bf16f32(a, bf, acc);
    }
    const float bias = db3[nt * 16 + nn];
    const float w    = fw[nt * 16 + nn];
#pragma unroll
    for (int v = 0; v < 8; ++v)
      partial[v] += sigf(acc[v] + bias) * w;
  }
#pragma unroll
  for (int v = 0; v < 8; ++v) {
    partial[v] += __shfl_xor(partial[v], 8, 32);
    partial[v] += __shfl_xor(partial[v], 4, 32);
    partial[v] += __shfl_xor(partial[v], 2, 32);
    partial[v] += __shfl_xor(partial[v], 1, 32);
  }
  if (nn == 0) {
    const float fbv = fb[0];
#pragma unroll
    for (int v = 0; v < 8; ++v) {
      const int r = r0 + v + hi;
      out[r] = partial[v] + fbv + item_bias_tab[item_id[r]];
    }
  }
}

// ---------------------------------------------------------------------------
// Launch
// ---------------------------------------------------------------------------
extern "C" void kernel_launch(void* const* d_in, const int* in_sizes, int n_in,
                              void* d_out, int out_size, void* d_ws, size_t ws_size,
                              hipStream_t stream) {
  (void)in_sizes; (void)n_in; (void)out_size; (void)ws_size;

  const int*   in_item_id  = (const int*)  d_in[0];
  const int*   item_id     = (const int*)  d_in[1];
  const float* item_emb    = (const float*)d_in[2];
  const float* item_bias   = (const float*)d_in[3];
  const float* aw1 = (const float*)d_in[4];  const float* ab1 = (const float*)d_in[5];
  const float* aw2 = (const float*)d_in[6];  const float* ab2 = (const float*)d_in[7];
  const float* afw = (const float*)d_in[8];  const float* afb = (const float*)d_in[9];
  const float* nw  = (const float*)d_in[10]; const float* nb  = (const float*)d_in[11];
  const float* dw1 = (const float*)d_in[12]; const float* db1 = (const float*)d_in[13];
  const float* dw2 = (const float*)d_in[14]; const float* db2 = (const float*)d_in[15];
  const float* dw3 = (const float*)d_in[16]; const float* db3 = (const float*)d_in[17];
  const float* fw  = (const float*)d_in[18]; const float* fb  = (const float*)d_in[19];

  char* ws = (char*)d_ws;
  size_t off = 0;
  auto alloc = [&](size_t bytes) -> void* {
    void* p = ws + off;
    off += (bytes + 63) & ~(size_t)63;
    return p;
  };
  __bf16* aw1p = (__bf16*)alloc((size_t)384 * 128 * 2);
  __bf16* aw2p = (__bf16*)alloc((size_t)128 * 64  * 2);
  __bf16* nwp  = (__bf16*)alloc((size_t)128 * 128 * 2);
  __bf16* dw1p = (__bf16*)alloc((size_t)384 * 512 * 2);
  __bf16* dw2p = (__bf16*)alloc((size_t)512 * 256 * 2);
  __bf16* dw3p = (__bf16*)alloc((size_t)256 * 128 * 2);
  float*  attn_ws = (float*)alloc((size_t)B_DIM * D_DIM * 4);

  auto pack = [&](const float* src, __bf16* dst, int K, int N) {
    int total = K * N;
    pack_b_kernel<<<(total + 255) / 256, 256, 0, stream>>>(src, dst, K, N, total);
  };
  pack(aw1, aw1p, 384, 128);
  pack(aw2, aw2p, 128, 64);
  pack(nw,  nwp,  128, 128);
  pack(dw1, dw1p, 384, 512);
  pack(dw2, dw2p, 512, 256);
  pack(dw3, dw3p, 256, 128);

  din_attn_kernel<<<B_DIM, 256, 0, stream>>>(
      in_item_id, item_id, item_emb, aw1p, ab1, aw2p, ab2, afw, afb, attn_ws);

  din_dnn_kernel<<<B_DIM / 64, 128, 0, stream>>>(
      attn_ws, item_id, item_emb, item_bias,
      nwp, nb, dw1p, db1, dw2p, db2, dw3p, db3, fw, fb, (float*)d_out);
}
